// OpenSERGE_18124761989301
// MI455X (gfx1250) — matrix operand order repaired and checked by reading, not compiled, not run on hardware
//
#include <hip/hip_runtime.h>

// ---------------------------------------------------------------------------
// MI455X (gfx1250) implementation.
//
//   K0 prep_weights : f32 -> f16 convert + transpose all GEMM weights (once)
//   K1 topk_nodes   : per-image bitonic sort of 16384 (key,idx) pairs in
//                     128 KB LDS -> exact top-256, node coords, valid mask
//   K2 gather       : x[N,C] = node_feats[:, topi] * valid ; g[d] degrees
//   K3 gnn_wmma     : whole 3-layer GNN for one image in ONE workgroup.
//                     Per layer, f16 operand tiles x16 and agg16 = g*(S-x)
//                     are materialized ONCE in LDS (2 x 135 KB, CDNA5-only),
//                     then each wave (fixed ni) runs k-outer / mi-inner
//                     WMMA with weight fragments hoisted out of the mi loop.
//   K4 edge_score   : per-edge hid = relu(A[s]+B[d]) ; logit = hid . Ws2.
// ---------------------------------------------------------------------------

#define NB   4
#define CF   256
#define HPIX 128
#define WPIX 128
#define NPIX (HPIX * WPIX)
#define NN   256                 // max_nodes
#define NL   3
#define SH   128                 // scorer hidden
#define NE   (NN * (NN - 1))     // 65280
#define STRH 264                 // f16 LDS row stride (528 B: conflict-free, 16B-aligned)
#define EPADH 132
#define STRIDE_F 4.0f

typedef __attribute__((ext_vector_type(16))) _Float16 v16h;
typedef __attribute__((ext_vector_type(8)))  _Float16 v8h;
typedef __attribute__((ext_vector_type(8)))  float    v8f;

extern __shared__ __align__(16) char dynsmem[];

__device__ __forceinline__ v16h cat8(v8h lo, v8h hi) {
  return __builtin_shufflevector(lo, hi, 0, 1, 2, 3, 4, 5, 6, 7,
                                         8, 9, 10, 11, 12, 13, 14, 15);
}

// ---------------------------------------------------------------------------
// K0: convert + transpose weights to f16 once.
//   wh[mat][n][k], mat = l*2+{self,nbr}; then scorer w1a[n][k], w1b[n][k].
// Transposed layout makes the WMMA B-fragment (K contiguous per lane) a
// single aligned 32-byte load.
// ---------------------------------------------------------------------------
__global__ void prep_weights_kernel(const float* __restrict__ Wself,
                                    const float* __restrict__ Wnbr,
                                    const float* __restrict__ Ws1,
                                    _Float16* __restrict__ wh) {
  int gid = blockIdx.x * blockDim.x + threadIdx.x;
  const int GNN_TOT = NL * 2 * CF * CF;   // 393216
  const int SC_TOT  = 2 * SH * CF;        // 65536
  if (gid < GNN_TOT) {
    int mat = gid >> 16;                  // CF*CF = 65536
    int e = gid & 65535;
    int n = e >> 8, k = e & 255;
    int l = mat >> 1, which = mat & 1;
    float s = which ? Wnbr[(size_t)l * CF * CF + k * CF + n]
                    : Wself[(size_t)l * CF * CF + k * CF + n];
    wh[(size_t)mat * CF * CF + n * CF + k] = (_Float16)s;
  } else if (gid < GNN_TOT + SC_TOT) {
    int e = gid - GNN_TOT;
    int which = e >> 15;                  // SH*CF = 32768
    int r = e & 32767;
    int n = r >> 8, k = r & 255;          // n < 128
    float s = Ws1[((size_t)which * CF + k) * SH + n];
    wh[(size_t)GNN_TOT + (size_t)which * SH * CF + n * CF + k] = (_Float16)s;
  }
}

// ---------------------------------------------------------------------------
// K1: exact top-256 via full bitonic sort in LDS (16384 * 8 B = 128 KB).
// Key = (~sortable(logit) << 32) | idx : ascending sort => logits descending,
// ties broken by smaller index (matches jax.lax.top_k).
// ---------------------------------------------------------------------------
__global__ void topk_nodes_kernel(const float* __restrict__ jl,
                                  const float* __restrict__ off,
                                  float* __restrict__ out_nodes,
                                  int* __restrict__ topi,
                                  float* __restrict__ validf) {
  const int b = blockIdx.x;
  const int tid = threadIdx.x;
  unsigned long long* keys = (unsigned long long*)dynsmem;
  for (int i = tid; i < NPIX; i += blockDim.x) {
    float f = jl[b * NPIX + i];
    unsigned u = __float_as_uint(f);
    u = (u & 0x80000000u) ? ~u : (u | 0x80000000u);   // ascending-sortable
    u = ~u;                                           // descending
    keys[i] = ((unsigned long long)u << 32) | (unsigned)i;
  }
  __syncthreads();
  for (int k = 2; k <= NPIX; k <<= 1) {
    for (int j = k >> 1; j > 0; j >>= 1) {
      for (int i = tid; i < NPIX; i += blockDim.x) {
        int ixj = i ^ j;
        if (ixj > i) {
          unsigned long long a = keys[i], c = keys[ixj];
          bool up = ((i & k) == 0);
          if ((a > c) == up) { keys[i] = c; keys[ixj] = a; }
        }
      }
      __syncthreads();
    }
  }
  if (tid < NN) {
    int pix = (int)(keys[tid] & 0xFFFFFFFFull);
    float lg = jl[b * NPIX + pix];
    validf[b * NN + tid] = (lg > 0.0f) ? 1.0f : 0.0f;  // sigmoid(lg) > 0.5
    topi[b * NN + tid] = pix;
    int iy = pix / WPIX, ix = pix % WPIX;
    float yo = off[(size_t)b * 2 * NPIX + 0 * NPIX + pix];
    float xo = off[(size_t)b * 2 * NPIX + 1 * NPIX + pix];
    out_nodes[((size_t)b * NN + tid) * 2 + 0] = ((float)ix + 0.5f + xo) * STRIDE_F;
    out_nodes[((size_t)b * NN + tid) * 2 + 1] = ((float)iy + 0.5f + yo) * STRIDE_F;
  }
}

// ---------------------------------------------------------------------------
// K2: gather x[N,C] (masked) + degree factors g[d] = valid[d]/(nvalid-1).
// ---------------------------------------------------------------------------
__global__ void gather_kernel(const float* __restrict__ nmap,
                              const int* __restrict__ topi,
                              const float* __restrict__ validf,
                              float* __restrict__ Xg,
                              float* __restrict__ gdeg) {
  const int b = blockIdx.x;
  const int t = threadIdx.x;   // 256 threads; t = channel
  __shared__ float sv[NN];
  __shared__ float nvs;
  sv[t] = validf[b * NN + t];
  __syncthreads();
  if (t == 0) {
    float nv = 0.f;
    for (int i = 0; i < NN; ++i) nv += sv[i];
    nvs = nv;
  }
  __syncthreads();
  float nv = nvs;
  gdeg[b * NN + t] = (sv[t] > 0.f && nv > 1.5f) ? (1.0f / (nv - 1.0f)) : 0.0f;
  const float* ch = nmap + ((size_t)b * CF + t) * NPIX;
  for (int n = 0; n < NN; ++n) {
    int pix = topi[b * NN + n];
    Xg[((size_t)b * NN + n) * CF + t] = ch[pix] * sv[n];
  }
}

// ---------------------------------------------------------------------------
// K3: full GNN for one image in one workgroup (512 threads = 16 waves).
//
// LDS (273 KB, needs CDNA5's 320 KB WGP LDS):
//   x16  [256][264] f16  : current activations, K-contiguous rows
//   ag16 [256][264] f16  : g[m]*(S[k]-x[m][k])  (complete-graph aggregation)
//   Sl/gl/vl             : colsum, degree factor, valid mask (f32)
//
// WMMA fragment mapping used here (wave32; contiguous K, not the ISA §7.12.2 table order):
//   A 16x32 f16 : lane (grp=lane>>4, ln=lane&15) row M=ln,
//                 elem e -> K = k0+grp*16+e (same order as B)
//                 => two 16-B ds_load_b128 per fragment, zero conversion.
//   B 32x16 f16 : col N=ln, elem e -> K = k0+grp*16+e (one 32-B load)
//   C/D 16x16 f32: VGPR r -> row M = grp*8 + r, col N = ln
//
// Each wave has fixed ni = wave; mi runs in 2 chunks of 8 with k0 outer so
// weight fragments load once per k-step (8x fewer than per-tile loads).
// ---------------------------------------------------------------------------
__global__ void __launch_bounds__(512)
gnn_wmma_kernel(float* __restrict__ Xg,              // [B][N][C] scratch (in/out)
                const float* __restrict__ gdeg,
                const float* __restrict__ validf,
                const _Float16* __restrict__ wh,     // prepped weights
                const float* __restrict__ b_gnn,     // [NL][C]
                const float* __restrict__ bs1,       // [SH]
                float* __restrict__ out_emb,         // [B][N][C]
                float* __restrict__ Aemb,            // [B][N][SH]
                float* __restrict__ Bemb) {          // [B][N][SH]
  const int b = blockIdx.x;
  const int tid = threadIdx.x;
  const int lane = tid & 31, wave = tid >> 5;        // 16 waves
  const int grp = lane >> 4, ln = lane & 15;
  _Float16* x16 = (_Float16*)dynsmem;                // [NN][STRH]
  _Float16* ag16 = x16 + NN * STRH;                  // [NN][STRH]
  float* Sl = (float*)(ag16 + NN * STRH);            // [CF]
  float* gl = Sl + CF;                               // [NN]
  float* vl = gl + NN;                               // [NN]
  float* Xrow = Xg + (size_t)b * NN * CF;

  for (int i = tid; i < NN; i += blockDim.x) { gl[i] = gdeg[b * NN + i]; vl[i] = validf[b * NN + i]; }

  // ---- operand build: x16 <- Xrow ; S ; ag16 ----
  auto build_x16 = [&]() {
    for (int i = tid; i < NN * CF; i += 512)
      x16[(i >> 8) * STRH + (i & 255)] = (_Float16)Xrow[i];
  };
  auto build_S_ag16 = [&]() {
    if (tid < CF) {
      float s = 0.f;
      for (int m = 0; m < NN; ++m) s += (float)x16[m * STRH + tid];
      Sl[tid] = s;
    }
    __syncthreads();
    for (int i = tid; i < NN * CF; i += 512) {
      int m = i >> 8, k = i & 255;
      ag16[m * STRH + k] = (_Float16)(gl[m] * (Sl[k] - (float)x16[m * STRH + k]));
    }
  };

  build_x16();
  __syncthreads();
  build_S_ag16();
  __syncthreads();

  const int ni = wave;                               // fixed column block
  const int col = ni * 16 + ln;

  for (int l = 0; l < NL; ++l) {
    const _Float16* wsel = wh + (size_t)(l * 2 + 0) * CF * CF + (size_t)col * CF;
    const _Float16* wnb  = wh + (size_t)(l * 2 + 1) * CF * CF + (size_t)col * CF;
    float bias = b_gnn[l * CF + col];
    for (int mh = 0; mh < 2; ++mh) {                 // 8 accumulators / chunk
      v8f acc[8];
      #pragma unroll
      for (int j = 0; j < 8; ++j) acc[j] = (v8f){};
      for (int k0 = 0; k0 < CF; k0 += 32) {
        int kb = k0 + grp * 16;
        v16h bsf = *(const v16h*)(wsel + kb);        // weight frags: 1x / k-step
        v16h bnf = *(const v16h*)(wnb + kb);
        __builtin_prefetch(wsel + kb + 32, 0, 0);
        __builtin_prefetch(wnb + kb + 32, 0, 0);
        int ka = k0 + grp * 16;
        #pragma unroll
        for (int j = 0; j < 8; ++j) {
          int m = (mh * 8 + j) * 16 + ln;
          const _Float16* xr = x16 + m * STRH + ka;
          const _Float16* ar = ag16 + m * STRH + ka;
          v16h ax = cat8(*(const v8h*)xr, *(const v8h*)(xr + 8));
          v16h ag = cat8(*(const v8h*)ar, *(const v8h*)(ar + 8));
          acc[j] = __builtin_amdgcn_wmma_f32_16x16x32_f16(false, ax, false, bsf,
                                                          (short)0, acc[j], false, false);
          acc[j] = __builtin_amdgcn_wmma_f32_16x16x32_f16(false, ag, false, bnf,
                                                          (short)0, acc[j], false, false);
        }
      }
      #pragma unroll
      for (int j = 0; j < 8; ++j) {
        #pragma unroll
        for (int r = 0; r < 8; ++r) {
          int row = (mh * 8 + j) * 16 + grp * 8 + r;
          float v = fmaxf(acc[j][r] + bias, 0.f) * vl[row];
          Xrow[(size_t)row * CF + col] = v;          // f32 master via scratch
        }
      }
    }
    __threadfence();
    __syncthreads();
    build_x16();
    __syncthreads();
    if (l < NL - 1) {
      build_S_ag16();
      __syncthreads();
    }
  }

  for (int i = tid; i < NN * CF; i += 512)
    out_emb[(size_t)b * NN * CF + i] = Xrow[i];

  // ---- factored scorer projections: A = x@Ws1_top + bs1 ; B = x@Ws1_bot ----
  // waves 0-7 -> A (ni2 = wave), waves 8-15 -> B (ni2 = wave-8); 16 mi each.
  {
    const int isB = wave >> 3;
    const int ni2 = wave & 7;
    const int col2 = ni2 * 16 + ln;
    const _Float16* wp = wh + (size_t)NL * 2 * CF * CF
                            + (size_t)isB * SH * CF + (size_t)col2 * CF;
    float bias = isB ? 0.f : bs1[col2];
    float* dst = isB ? Bemb : Aemb;
    for (int mh = 0; mh < 2; ++mh) {
      v8f acc[8];
      #pragma unroll
      for (int j = 0; j < 8; ++j) acc[j] = (v8f){};
      for (int k0 = 0; k0 < CF; k0 += 32) {
        v16h bf = *(const v16h*)(wp + k0 + grp * 16);
        int ka = k0 + grp * 16;
        #pragma unroll
        for (int j = 0; j < 8; ++j) {
          int m = (mh * 8 + j) * 16 + ln;
          const _Float16* xr = x16 + m * STRH + ka;
          v16h ax = cat8(*(const v8h*)xr, *(const v8h*)(xr + 8));
          acc[j] = __builtin_amdgcn_wmma_f32_16x16x32_f16(false, ax, false, bf,
                                                          (short)0, acc[j], false, false);
        }
      }
      #pragma unroll
      for (int j = 0; j < 8; ++j) {
        #pragma unroll
        for (int r = 0; r < 8; ++r) {
          int row = (mh * 8 + j) * 16 + grp * 8 + r;
          dst[((size_t)b * NN + row) * SH + col2] = acc[j][r] + bias;
        }
      }
    }
  }
}

// ---------------------------------------------------------------------------
// K4: per-edge score. Block = (chunk of 8 src nodes, image); whole B matrix
// for the image staged once in LDS (256x132 f32 = 135 KB, padded stride so
// the 32 lanes' different d rows hit distinct banks).
// Edge order matches reference: e = s*(N-1) + t ; d = t<s ? t : t+1.
// ---------------------------------------------------------------------------
__global__ void edge_score_kernel(const float* __restrict__ Aemb,
                                  const float* __restrict__ Bemb,
                                  const float* __restrict__ validf,
                                  const float* __restrict__ Ws2,
                                  const float* __restrict__ bs2p,
                                  float* __restrict__ out_logits,
                                  float* __restrict__ out_keep) {
  const int chunk = blockIdx.x;    // 0..31
  const int b = blockIdx.y;
  const int tid = threadIdx.x;     // 256
  float* Bt = (float*)dynsmem;                 // [NN][EPADH]
  float* At = Bt + NN * EPADH;                 // [8][EPADH]
  float* w2 = At + 8 * EPADH;                  // [SH]
  for (int i = tid; i < NN * SH; i += 256) {
    int d = i >> 7, h = i & 127;
    Bt[d * EPADH + h] = Bemb[((size_t)b * NN + d) * SH + h];
  }
  for (int i = tid; i < 8 * SH; i += 256) {
    int sl = i >> 7, h = i & 127;
    At[sl * EPADH + h] = Aemb[((size_t)b * NN + chunk * 8 + sl) * SH + h];
  }
  if (tid < SH) w2[tid] = Ws2[tid];
  __syncthreads();
  float bs2 = bs2p[0];
  for (int sl = 0; sl < 8; ++sl) {
    int s = chunk * 8 + sl;
    if (tid < NN - 1) {
      int d = (tid < s) ? tid : tid + 1;
      float acc = 0.f;
      #pragma unroll 4
      for (int h = 0; h < SH; ++h)
        acc += fmaxf(At[sl * EPADH + h] + Bt[d * EPADH + h], 0.f) * w2[h];
      float logit = acc + bs2;
      size_t e = (size_t)b * NE + (size_t)s * (NN - 1) + tid;
      out_logits[e] = logit;
      bool kp = (logit > 0.f) && (validf[b * NN + s] > 0.f) && (validf[b * NN + d] > 0.f);
      out_keep[e] = kp ? 1.0f : 0.0f;
    }
  }
}

// ---------------------------------------------------------------------------
extern "C" void kernel_launch(void* const* d_in, const int* in_sizes, int n_in,
                              void* d_out, int out_size, void* d_ws, size_t ws_size,
                              hipStream_t stream) {
  (void)in_sizes; (void)n_in; (void)out_size; (void)ws_size;
  const float* jl    = (const float*)d_in[0];
  const float* off   = (const float*)d_in[1];
  const float* nmap  = (const float*)d_in[2];
  const float* Wself = (const float*)d_in[3];
  const float* Wnbr  = (const float*)d_in[4];
  const float* bgnn  = (const float*)d_in[5];
  const float* Ws1   = (const float*)d_in[6];
  const float* bs1   = (const float*)d_in[7];
  const float* Ws2   = (const float*)d_in[8];
  const float* bs2   = (const float*)d_in[9];

  char* ws = (char*)d_ws;
  int*      topi   = (int*)(ws + 0);                       //   4 KB
  float*    validf = (float*)(ws + 4096);                  //   4 KB
  float*    gdeg   = (float*)(ws + 8192);                  //   4 KB
  float*    Xg     = (float*)(ws + 16384);                 //   1 MB
  float*    Aemb   = (float*)(ws + 16384 + 1048576);       // 512 KB
  float*    Bemb   = (float*)(ws + 16384 + 1048576 + 524288);            // 512 KB
  _Float16* wh     = (_Float16*)(ws + 16384 + 1048576 + 2 * 524288);     // 896 KB

  float* out_nodes  = (float*)d_out;                       // [4,256,2]
  float* out_emb    = out_nodes + NB * NN * 2;             // [4,256,256]
  float* out_logits = out_emb + (size_t)NB * NN * CF;      // [4,65280]
  float* out_keep   = out_logits + (size_t)NB * NE;        // [4,65280]

  prep_weights_kernel<<<(NL * 2 * CF * CF + 2 * SH * CF + 255) / 256, 256, 0, stream>>>(
      Wself, Wnbr, Ws1, wh);
  topk_nodes_kernel<<<NB, 1024, NPIX * sizeof(unsigned long long), stream>>>(
      jl, off, out_nodes, topi, validf);
  gather_kernel<<<NB, NN, 0, stream>>>(nmap, topi, validf, Xg, gdeg);
  size_t smem3 = (size_t)(2 * NN * STRH) * sizeof(_Float16)
               + (size_t)(CF + NN + NN) * sizeof(float);                  // ~273 KB
  gnn_wmma_kernel<<<NB, 512, smem3, stream>>>(Xg, gdeg, validf, wh, bgnn, bs1,
                                              out_emb, Aemb, Bemb);
  size_t smem4 = (size_t)(NN * EPADH + 8 * EPADH + SH) * sizeof(float);   // ~137 KB
  edge_score_kernel<<<dim3(32, NB), 256, smem4, stream>>>(
      Aemb, Bemb, validf, Ws2, bs2, out_logits, out_keep);
}
